// AttentionWeightedValues_71631464563513
// MI455X (gfx1250) — 8-bit matrix operand order repaired and checked by reading, compiled, not run on hardware
//
#include <hip/hip_runtime.h>
#include <hip/hip_bf16.h>

constexpr int Bc = 2, Hc = 16, Tc = 2048, Sc = 2048, Dc = 128;
constexpr int BH = Bc * Hc;

typedef __attribute__((ext_vector_type(16))) int   v16i;
typedef __attribute__((ext_vector_type(8)))  float v8f;

#if __has_builtin(__builtin_amdgcn_global_load_async_to_lds_b128) && \
    __has_builtin(__builtin_amdgcn_s_wait_asynccnt)
#define HAVE_ASYNC_LDS 1
typedef int v4i_vs __attribute__((vector_size(16)));
typedef __attribute__((address_space(1))) v4i_vs* as1_v4i;
typedef __attribute__((address_space(3))) v4i_vs* as3_v4i;
#endif

#if __has_builtin(__builtin_amdgcn_cvt_pk_fp8_f32)
__device__ __forceinline__ unsigned pack4_e4m3(float x0, float x1, float x2, float x3) {
  int p = 0;
  p = __builtin_amdgcn_cvt_pk_fp8_f32(x0, x1, p, false);
  p = __builtin_amdgcn_cvt_pk_fp8_f32(x2, x3, p, true);
  return (unsigned)p;
}
#else
__device__ __forceinline__ unsigned f32_to_e4m3_1(float x) {
  unsigned u = __float_as_uint(x);
  unsigned s = (u >> 24) & 0x80u;
  float ax = __uint_as_float(u & 0x7fffffffu);
  if (!(ax == ax)) return s | 0x7fu;
  ax = fminf(ax, 448.0f);
  if (ax < 0x1.0p-6f) {
    unsigned m = (unsigned)rintf(ax * 512.0f);
    return s | m;
  }
  unsigned b = __float_as_uint(ax);
  unsigned mant = b & 0x7fffffu;
  unsigned keep = mant >> 20;
  unsigned rest = mant & 0xfffffu;
  keep += (rest > 0x80000u) || (rest == 0x80000u && (keep & 1u));
  int e = (int)((b >> 23) & 0xffu) - 127;
  if (keep == 8u) { keep = 0u; e += 1; }
  if (e > 8) { e = 8; keep = 6u; }
  return s | ((unsigned)(e + 7) << 3) | keep;
}
__device__ __forceinline__ unsigned pack4_e4m3(float x0, float x1, float x2, float x3) {
  return f32_to_e4m3_1(x0) | (f32_to_e4m3_1(x1) << 8) |
         (f32_to_e4m3_1(x2) << 16) | (f32_to_e4m3_1(x3) << 24);
}
#endif

__device__ __forceinline__ float clamp448(float x) {
  return fminf(fmaxf(x, -448.0f), 448.0f);
}

__global__ void init_ws_kernel(unsigned* u) {
  if (threadIdx.x < 2) u[threadIdx.x] = 0u;
}

__global__ __launch_bounds__(256) void amax_kernel(const float* __restrict__ x,
                                                   size_t n4,
                                                   unsigned* __restrict__ dst) {
  size_t i = (size_t)blockIdx.x * blockDim.x + threadIdx.x;
  size_t stride = (size_t)gridDim.x * blockDim.x;
  const float4* x4 = (const float4*)x;
  float m = 0.0f;
  for (; i < n4; i += stride) {
    float4 f = x4[i];
    m = fmaxf(m, fmaxf(fmaxf(fabsf(f.x), fabsf(f.y)), fmaxf(fabsf(f.z), fabsf(f.w))));
  }
  #pragma unroll
  for (int off = 16; off > 0; off >>= 1)
    m = fmaxf(m, __shfl_xor(m, off, 32));
  if ((threadIdx.x & 31) == 0)
    atomicMax(dst, __float_as_uint(m));
}

__global__ __launch_bounds__(256) void quantT_v_kernel(const float* __restrict__ v,
                                                       unsigned char* __restrict__ vt,
                                                       const unsigned* __restrict__ wsu) {
  __shared__ unsigned char Tl[64][68];
  const int s0 = blockIdx.x * 64;
  const int d0 = blockIdx.y * 64;
  const int bh = blockIdx.z;
  const float amax_b = fmaxf(__uint_as_float(wsu[1]), 1e-12f);
  const float inv_sb = 448.0f / amax_b;

  const float* vb = v + ((size_t)bh * Sc + s0) * Dc + d0;
  float4 fst[4];
  #pragma unroll
  for (int it = 0; it < 4; ++it) {
    int idx = it * 256 + threadIdx.x;
    int r = idx >> 4;
    int c4 = idx & 15;
    fst[it] = ((const float4*)(vb + (size_t)r * Dc))[c4];
  }
  #pragma unroll
  for (int it = 0; it < 4; ++it) {
    int idx = it * 256 + threadIdx.x;
    int r = idx >> 4;
    int c4 = idx & 15;
    float4 f = fst[it];
    unsigned u = pack4_e4m3(clamp448(f.x * inv_sb), clamp448(f.y * inv_sb),
                            clamp448(f.z * inv_sb), clamp448(f.w * inv_sb));
    *(unsigned*)&Tl[r][c4 * 4] = u;
  }
  __syncthreads();
  unsigned char* ob = vt + ((size_t)bh * Dc + d0) * Sc + s0;
  #pragma unroll
  for (int it = 0; it < 4; ++it) {
    int idx = it * 256 + threadIdx.x;
    int dr = idx >> 4;
    int su = idx & 15;
    unsigned u = (unsigned)Tl[su * 4 + 0][dr] |
                 ((unsigned)Tl[su * 4 + 1][dr] << 8) |
                 ((unsigned)Tl[su * 4 + 2][dr] << 16) |
                 ((unsigned)Tl[su * 4 + 3][dr] << 24);
    *(unsigned*)(ob + (size_t)dr * Sc + su * 4) = u;
  }
}

constexpr int TM = 128;
constexpr int TK = 128;
constexpr int LDP = 144;

__global__ __launch_bounds__(256) void fused_fp8_gemm_kernel(
    const float* __restrict__ aw, const unsigned char* __restrict__ vt,
    const unsigned* __restrict__ wsu, float* __restrict__ out) {
  __shared__ unsigned char Alds[TM][LDP];
  __shared__ unsigned char Blds[Dc][LDP];

  const int tid  = threadIdx.x;
  const int lane = tid & 31;
  const int wave = tid >> 5;
  const int bh = blockIdx.y;
  const int b  = bh >> 4;
  const int h  = bh & 15;
  const int m0 = blockIdx.x * TM;

  const float amax_a = fmaxf(__uint_as_float(wsu[0]), 1e-12f);
  const float amax_b = fmaxf(__uint_as_float(wsu[1]), 1e-12f);
  const float inv_sa = 448.0f / amax_a;
  const float out_scale = (amax_a / 448.0f) * (amax_b / 448.0f);

  const float*         Ab = aw + ((size_t)bh * Tc + m0) * Sc;
  const unsigned char* Bb = vt + (size_t)bh * Dc * Sc;

  v8f acc[8] = {};

  const int mrow = lane & 15;
  const int hi   = lane >> 4;

  for (int kk = 0; kk < Sc; kk += TK) {
    __syncthreads();

#ifdef HAVE_ASYNC_LDS
    #pragma unroll
    for (int it = 0; it < 4; ++it) {
      int idx = it * 256 + tid;
      int r   = idx >> 3;
      int c16 = idx & 7;
      __builtin_amdgcn_global_load_async_to_lds_b128(
          (as1_v4i)(void*)(const_cast<unsigned char*>(Bb) + (size_t)r * Sc + kk + c16 * 16),
          (as3_v4i)(void*)&Blds[r][c16 * 16], 0, 0);
    }
#else
    int4 bq[4];
    #pragma unroll
    for (int it = 0; it < 4; ++it) {
      int idx = it * 256 + tid;
      int r   = idx >> 3;
      int c16 = idx & 7;
      bq[it] = *(const int4*)(Bb + (size_t)r * Sc + kk + c16 * 16);
    }
#endif

    if (kk + TK < Sc) {
      #pragma unroll
      for (int it = 0; it < 4; ++it) {
        int idx = it * 256 + tid;
        int r = idx >> 3;
        int l = idx & 7;
        __builtin_prefetch(Ab + (size_t)r * Sc + kk + TK + l * 16, 0, 1);
      }
    }

    #pragma unroll
    for (int bt = 0; bt < 2; ++bt) {
      float4 st[8];
      #pragma unroll
      for (int it = 0; it < 8; ++it) {
        int idx = (bt * 8 + it) * 256 + tid;
        int r   = idx >> 5;
        int c4  = idx & 31;
        st[it] = ((const float4*)(Ab + (size_t)r * Sc + kk))[c4];
      }
      #pragma unroll
      for (int it = 0; it < 8; ++it) {
        int idx = (bt * 8 + it) * 256 + tid;
        int r   = idx >> 5;
        int c4  = idx & 31;
        float4 f = st[it];
        unsigned u = pack4_e4m3(clamp448(f.x * inv_sa), clamp448(f.y * inv_sa),
                                clamp448(f.z * inv_sa), clamp448(f.w * inv_sa));
        *(unsigned*)&Alds[r][c4 * 4] = u;
      }
    }

#ifdef HAVE_ASYNC_LDS
    __builtin_amdgcn_s_wait_asynccnt(0);
#else
    #pragma unroll
    for (int it = 0; it < 4; ++it) {
      int idx = it * 256 + tid;
      int r   = idx >> 3;
      int c16 = idx & 7;
      *(int4*)&Blds[r][c16 * 16] = bq[it];
    }
#endif
    __syncthreads();

    v16i a;
    const unsigned char* ar = &Alds[wave * 16 + mrow][0];
    #pragma unroll
    for (int half = 0; half < 2; ++half) {
      #pragma unroll
      for (int c = 0; c < 4; ++c) {
        int2 d2 = *(const int2*)(ar + half * 64 + (c >> 1) * 32 + (c & 1) * 8 + hi * 16);
        a[half * 8 + c * 2]     = d2.x;
        a[half * 8 + c * 2 + 1] = d2.y;
      }
    }

    #pragma unroll
    for (int nt = 0; nt < 8; ++nt) {
      v16i bf;
      const unsigned char* br = &Blds[nt * 16 + mrow][0];
      #pragma unroll
      for (int c = 0; c < 4; ++c) {
        int4 q = *(const int4*)(br + hi * 16 + c * 32);
        bf[4 * c]     = q.x;
        bf[4 * c + 1] = q.y;
        bf[4 * c + 2] = q.z;
        bf[4 * c + 3] = q.w;
      }
      acc[nt] = __builtin_amdgcn_wmma_f32_16x16x128_fp8_fp8(
          a, bf, (short)0, acc[nt], false, false);
    }
  }

  #pragma unroll
  for (int nt = 0; nt < 8; ++nt) {
    #pragma unroll
    for (int r = 0; r < 8; ++r) {
      int m = m0 + wave * 16 + hi * 8 + r;
      int n = nt * 16 + mrow;
      out[(((size_t)b * Tc + m) * Hc + h) * Dc + n] = acc[nt][r] * out_scale;
    }
  }
}

extern "C" void kernel_launch(void* const* d_in, const int* in_sizes, int n_in,
                              void* d_out, int out_size, void* d_ws, size_t ws_size,
                              hipStream_t stream) {
  (void)in_sizes; (void)n_in; (void)out_size; (void)ws_size;
  const float* aw = (const float*)d_in[0];
  const float* v  = (const float*)d_in[1];
  float* out = (float*)d_out;

  unsigned* wsu = (unsigned*)d_ws;
  unsigned char* vt = (unsigned char*)d_ws + 256;

  init_ws_kernel<<<1, 32, 0, stream>>>(wsu);

  amax_kernel<<<4096, 256, 0, stream>>>(aw, (size_t)BH * Tc * Sc / 4, wsu + 0);
  amax_kernel<<<512,  256, 0, stream>>>(v,  (size_t)BH * Sc * Dc / 4, wsu + 1);

  quantT_v_kernel<<<dim3(Sc / 64, Dc / 64, BH), 256, 0, stream>>>(v, vt, wsu);

  fused_fp8_gemm_kernel<<<dim3(Tc / TM, BH), 256, 0, stream>>>(aw, vt, wsu, out);
}
